// ConditionedPNA_26207890440681
// MI455X (gfx1250) — matrix operand order repaired and checked by reading, not compiled, not run on hardware
//
#include <hip/hip_runtime.h>
#include <cmath>

#define FD 128            // feature dim
#define KPRE 256          // GEMM1 output cols (u|z)
#define KPOST 640         // GEMM2 K (h 128 + agg 512)

typedef __attribute__((ext_vector_type(16))) __bf16 v16bf;
typedef __attribute__((ext_vector_type(8)))  float  v8f;

// Native f32 -> bf16 (compiler lowers to gfx1250 hardware cvt; avoids the
// VALU bit-twiddle chains that blew up register pressure in round 1).
__device__ __forceinline__ __bf16 f2bf(float f) { return (__bf16)f; }

__device__ __forceinline__ v16bf load_a_frag(const float* __restrict__ ar) {
  // ar already offset by k0 + khalf; elements j -> K = khalf+j (contiguous 16)
  float4 p0 = *(const float4*)(ar);
  float4 p1 = *(const float4*)(ar + 4);
  float4 p2 = *(const float4*)(ar + 8);
  float4 p3 = *(const float4*)(ar + 12);
  v16bf a;
  a[0]=f2bf(p0.x);  a[1]=f2bf(p0.y);  a[2]=f2bf(p0.z);  a[3]=f2bf(p0.w);
  a[4]=f2bf(p1.x);  a[5]=f2bf(p1.y);  a[6]=f2bf(p1.z);  a[7]=f2bf(p1.w);
  a[8]=f2bf(p2.x);  a[9]=f2bf(p2.y);  a[10]=f2bf(p2.z); a[11]=f2bf(p2.w);
  a[12]=f2bf(p3.x); a[13]=f2bf(p3.y); a[14]=f2bf(p3.z); a[15]=f2bf(p3.w);
  return a;
}

// ---------------------------------------------------------------------------
// Weight folding: W2[p,j] = sum_k Wlin[p,k] * Weff[k,j]
// Weff[k,j] = Wpost[k,j]                                    (j < 128, "h" block)
//           = Wpost[k,j] + c1*Wpost[k,j+512] + c2*Wpost[k,j+1024]  (j in [128,640))
// ---------------------------------------------------------------------------
__global__ void prep_w2(const float* __restrict__ Wpost,   // (128, 1664)
                        const float* __restrict__ Wlin,    // (128, 128)
                        float c1, float c2,
                        float* __restrict__ W2)            // (128, 640)
{
  int idx = blockIdx.x * blockDim.x + threadIdx.x;
  if (idx >= FD * KPOST) return;
  int p = idx / KPOST, j = idx % KPOST;
  float acc = 0.0f;
  for (int k = 0; k < FD; ++k) {
    float w = Wpost[k * (13 * FD) + j];
    if (j >= FD)
      w += c1 * Wpost[k * (13 * FD) + j + 512] + c2 * Wpost[k * (13 * FD) + j + 1024];
    acc += Wlin[p * FD + k] * w;
  }
  W2[p * KPOST + j] = acc;
}

__global__ void prep_b2(const float* __restrict__ Wlin, const float* __restrict__ bpost,
                        const float* __restrict__ blin, float* __restrict__ b2)
{
  int p = threadIdx.x;
  float acc = blin[p];
  for (int k = 0; k < FD; ++k) acc += Wlin[p * FD + k] * bpost[k];
  b2[p] = acc;
}

// ---------------------------------------------------------------------------
// GEMM1: uz(N,256) = h(N,128) @ Bmat(128,256), Bmat[k,n] = n<128 ? Wpre[n,k]
//                                                         : Wpre[n-128,128+k]
// u part (n<128) gets +bpre.
// Block = 8 waves: waves 0-3 -> 4 row-tiles x cols[0,128); waves 4-7 -> same
// row-tiles x cols[128,256). 8 C-frags per wave (64 VGPRs) => no spills.
// ---------------------------------------------------------------------------
__global__ void pna_gemm_pre(const float* __restrict__ h,
                             const float* __restrict__ Wpre,   // (128, 256)
                             const float* __restrict__ bpre,   // (128)
                             float* __restrict__ uz, int nrows)
{
  extern __shared__ __bf16 Bs[];            // [256 cols][32 k], K-contiguous
  const int tid  = threadIdx.x;
  const int lane = tid & 31, wave = tid >> 5;
  const int row0    = (blockIdx.x * 4 + (wave & 3)) * 16;
  const int colBase = (wave >> 2) * FD;     // 0 (u half) or 128 (z half)
  const int khalf = (lane < 16) ? 0 : 16;   // A-fragment K split
  const int bk    = (lane < 16) ? 0 : 16;   // B-fragment K split
  const int col0  = lane & 15;
  int am = row0 + (lane & 15);
  if (am > nrows - 1) am = nrows - 1;       // clamp: compute garbage, skip store
  const float* arow = h + (size_t)am * FD;

  v8f acc[8] = {};
  for (int k0 = 0; k0 < FD; k0 += 32) {
    __syncthreads();
    for (int idx = tid; idx < 32 * KPRE; idx += 256) {
      int kk = idx & 31, n = idx >> 5;
      float w = (n < FD) ? Wpre[n * 256 + (k0 + kk)]
                         : Wpre[(n - FD) * 256 + FD + (k0 + kk)];
      Bs[n * 32 + kk] = f2bf(w);
    }
    __syncthreads();
    v16bf a = load_a_frag(arow + k0 + khalf);
#pragma unroll
    for (int t = 0; t < 8; ++t) {
      v16bf b = *(const v16bf*)(Bs + ((colBase + t * 16 + col0) * 32 + bk));
      acc[t] = __builtin_amdgcn_wmma_f32_16x16x32_bf16(
                   false, a, false, b, (short)0, acc[t], false, false);
    }
  }
  const int rbase = row0 + ((lane < 16) ? 0 : 8);
#pragma unroll
  for (int t = 0; t < 8; ++t) {
    int c = colBase + t * 16 + col0;
    float bias = (c < FD) ? bpre[c] : 0.0f;
#pragma unroll
    for (int v = 0; v < 8; ++v) {
      int r = rbase + v;
      if (r < nrows) uz[(size_t)r * KPRE + c] = acc[t][v] + bias;
    }
  }
}

// ---------------------------------------------------------------------------
// Aggregate: one wave per node, 4 channels per lane. m_e = u_i + z_{col[e]}.
// agg row = [mean | max | min | std], (N, 512). uz (51 MB) is L2-resident.
// ---------------------------------------------------------------------------
__global__ void pna_aggregate(const float* __restrict__ uz,
                              const int* __restrict__ col,
                              float* __restrict__ agg,
                              int n, int deg, float inv_deg)
{
  const int node = blockIdx.x * 8 + (threadIdx.x >> 5);
  if (node >= n) return;
  const int lane = threadIdx.x & 31;
  const int ch = lane * 4;

  float uu[4];
  { float4 t = *(const float4*)(uz + (size_t)node * KPRE + ch);
    uu[0]=t.x; uu[1]=t.y; uu[2]=t.z; uu[3]=t.w; }

  float s[4]  = {0,0,0,0}, s2[4] = {0,0,0,0};
  float mx[4] = {-3.4e38f,-3.4e38f,-3.4e38f,-3.4e38f};
  float mn[4] = { 3.4e38f, 3.4e38f, 3.4e38f, 3.4e38f};

  const int* ce = col + (size_t)node * deg;
  for (int e = 0; e < deg; ++e) {
    int c = ce[e];
    if (e + 1 < deg)
      __builtin_prefetch(uz + (size_t)ce[e + 1] * KPRE + FD + ch, 0, 0);
    float4 zt = *(const float4*)(uz + (size_t)c * KPRE + FD + ch);
    float zz[4] = {zt.x, zt.y, zt.z, zt.w};
#pragma unroll
    for (int j = 0; j < 4; ++j) {
      float m = uu[j] + zz[j];
      s[j]  += m;
      s2[j] += m * m;
      mx[j]  = fmaxf(mx[j], m);
      mn[j]  = fminf(mn[j], m);
    }
  }
  float me[4], sd[4];
#pragma unroll
  for (int j = 0; j < 4; ++j) {
    me[j] = s[j] * inv_deg;
    float var = s2[j] * inv_deg - me[j] * me[j];
    sd[j] = sqrtf(fmaxf(var, 0.0f) + 1e-5f);
  }
  float* ag = agg + (size_t)node * 512 + ch;
  *(float4*)(ag)       = make_float4(me[0], me[1], me[2], me[3]);
  *(float4*)(ag + 128) = make_float4(mx[0], mx[1], mx[2], mx[3]);
  *(float4*)(ag + 256) = make_float4(mn[0], mn[1], mn[2], mn[3]);
  *(float4*)(ag + 384) = make_float4(sd[0], sd[1], sd[2], sd[3]);
}

// ---------------------------------------------------------------------------
// GEMM2: out(N,128) = [h | agg](N,640) @ W2.T + b2 + h   (bias+residual fused)
// Block = 8 waves x (16 rows x 128 cols); 8 C-frags per wave.
// ---------------------------------------------------------------------------
__global__ void pna_gemm_post(const float* __restrict__ h,
                              const float* __restrict__ agg,  // (N, 512)
                              const float* __restrict__ W2,   // (128, 640)
                              const float* __restrict__ b2,   // (128)
                              float* __restrict__ out, int nrows)
{
  extern __shared__ __bf16 Bs[];            // [128 cols][32 k]
  const int tid  = threadIdx.x;
  const int lane = tid & 31, wave = tid >> 5;
  const int row0  = (blockIdx.x * 8 + wave) * 16;
  const int khalf = (lane < 16) ? 0 : 16;
  const int bk    = (lane < 16) ? 0 : 16;
  const int col0  = lane & 15;
  int am = row0 + (lane & 15);
  if (am > nrows - 1) am = nrows - 1;
  const float* hrow = h   + (size_t)am * FD;
  const float* grow = agg + (size_t)am * 512;

  v8f acc[8] = {};
  for (int k0 = 0; k0 < KPOST; k0 += 32) {
    __syncthreads();
    for (int idx = tid; idx < 32 * FD; idx += 256) {
      int kk = idx & 31, n = idx >> 5;
      Bs[n * 32 + kk] = f2bf(W2[n * KPOST + k0 + kk]);
    }
    __syncthreads();
    const float* ar = (k0 < FD) ? (hrow + k0 + khalf)
                                : (grow + (k0 - FD) + khalf);
    v16bf a = load_a_frag(ar);
#pragma unroll
    for (int t = 0; t < 8; ++t) {
      v16bf b = *(const v16bf*)(Bs + ((t * 16 + col0) * 32 + bk));
      acc[t] = __builtin_amdgcn_wmma_f32_16x16x32_bf16(
                   false, a, false, b, (short)0, acc[t], false, false);
    }
  }
  const int rbase = row0 + ((lane < 16) ? 0 : 8);
#pragma unroll
  for (int t = 0; t < 8; ++t) {
    int c = t * 16 + col0;
    float bias = b2[c];
#pragma unroll
    for (int v = 0; v < 8; ++v) {
      int r = rbase + v;
      if (r < nrows)
        out[(size_t)r * FD + c] = acc[t][v] + bias + h[(size_t)r * FD + c];
    }
  }
}

// ---------------------------------------------------------------------------
extern "C" void kernel_launch(void* const* d_in, const int* in_sizes, int n_in,
                              void* d_out, int out_size, void* d_ws, size_t ws_size,
                              hipStream_t stream) {
  const float* x     = (const float*)d_in[0];
  const int*   eidx  = (const int*)  d_in[1];
  const float* Wpre  = (const float*)d_in[2];   // (3, 128, 256)
  const float* bpre  = (const float*)d_in[3];   // (3, 128)
  const float* Wpost = (const float*)d_in[4];   // (3, 128, 1664)
  const float* bpost = (const float*)d_in[5];   // (3, 128)
  const float* Wlin  = (const float*)d_in[6];   // (3, 128, 128)
  const float* blin  = (const float*)d_in[7];   // (3, 128)
  float* out = (float*)d_out;

  const int N   = in_sizes[0] / FD;
  const int E   = in_sizes[1] / 2;
  const int deg = E / N;                        // structurally constant (=12)
  const int* col = eidx + E;                    // edge_index[1]

  // scaler constants: dlog = log(deg+1) is the same for every node
  double sAvg = 0.0;
  for (int i = 0; i < 20; ++i) sAvg += (i < 10 ? 1.0 : 2.0) * log((double)i + 1.0);
  const float avg_log = (float)(sAvg / 30.0);
  const float dlog    = logf((float)deg + 1.0f);
  const float c1 = dlog / avg_log, c2 = avg_log / dlog;

  // workspace carve-up (floats)
  float* ws = (float*)d_ws;
  size_t off = 0;
  float* uz  = ws + off; off += (size_t)N * KPRE;   // (N,256)  u|z
  float* agg = ws + off; off += (size_t)N * 512;    // (N,512)  mean|max|min|std
  float* h1  = ws + off; off += (size_t)N * FD;
  float* h2  = ws + off; off += (size_t)N * FD;
  float* W2  = ws + off; off += (size_t)FD * KPOST;
  float* b2  = ws + off; off += FD;

  const int preBlocks  = (N + 63) / 64;    // 4 row-tiles of 16 per block
  const int postBlocks = (N + 127) / 128;  // 8 row-tiles of 16 per block
  const float* hin = x;
  for (int l = 0; l < 3; ++l) {
    float* hout = (l == 2) ? out : (l == 0 ? h1 : h2);
    const float* Wpre_l  = Wpre  + (size_t)l * FD * 256;
    const float* bpre_l  = bpre  + (size_t)l * FD;
    const float* Wpost_l = Wpost + (size_t)l * FD * (13 * FD);
    const float* bpost_l = bpost + (size_t)l * FD;
    const float* Wlin_l  = Wlin  + (size_t)l * FD * FD;
    const float* blin_l  = blin  + (size_t)l * FD;

    prep_w2<<<(FD * KPOST + 255) / 256, 256, 0, stream>>>(Wpost_l, Wlin_l, c1, c2, W2);
    prep_b2<<<1, FD, 0, stream>>>(Wlin_l, bpost_l, blin_l, b2);
    pna_gemm_pre<<<preBlocks, 256, 32 * KPRE * sizeof(__bf16), stream>>>(
        hin, Wpre_l, bpre_l, uz, N);
    pna_aggregate<<<(N + 7) / 8, 256, 0, stream>>>(uz, col, agg, N, deg, 1.0f / (float)deg);
    pna_gemm_post<<<postBlocks, 256, 32 * FD * sizeof(__bf16), stream>>>(
        hin, agg, W2, b2, hout, N);
    hin = hout;
  }
}
